// Extra_Modal_Modeling_19215683682434
// MI455X (gfx1250) — matrix operand order repaired and checked by reading, not compiled, not run on hardware
//
#include <hip/hip_runtime.h>
#include <math.h>
#include <stdint.h>

// ---------------------------------------------------------------------------
// MI455X (gfx1250) implementation.
//
// Precision strategy: all large GEMMs run as bf16 WMMA (v_wmma_f32_16x16x32_bf16,
// f32 accumulate). Everything else (LN, GELU(erf), softmax, top-k, small 8-row
// linears) stays f32 VALU. Roofline: ~1.4 TFLOP of GEMM vs ~2-3 GB HBM traffic;
// WMMA keeps us on the bandwidth side of the roofline (~100us floor @ 23.3TB/s).
//
// GEMM data movement: CDNA5 async global->LDS DMA (GLOBAL_LOAD_ASYNC_TO_LDS_B128,
// ASYNCcnt-tracked, double-buffered LDS) -- zero VGPRs spent on staging, which
// previously forced scratch spills.
// ---------------------------------------------------------------------------

#ifndef USE_ASYNC_LDS
#define USE_ASYNC_LDS 1     // flip to 0 if the async mnemonic fails to assemble
#endif

typedef __bf16 bf16;
typedef __attribute__((ext_vector_type(8)))  bf16  v8bf;
typedef __attribute__((ext_vector_type(16))) bf16  v16bf;
typedef __attribute__((ext_vector_type(8)))  float v8f;

struct U128 { unsigned u[4]; };
struct F4   { float x, y, z, w; };

// ---------------- constants ----------------
#define Dm   1024
#define NHm  8
#define TOKm 128
#define Bm   8
#define Nm   4096
#define Mm   4096
#define Hm   2048
#define RMAXm 32776   // B*(N+1)

// Generic LDS pointer -> wave-relative LDS byte offset (ISA 10.2: addr[31:0]).
__device__ __forceinline__ unsigned lds_off(const void* p)
{
    return (unsigned)(uintptr_t)p;
}

#if defined(__has_builtin)
#if __has_builtin(__builtin_amdgcn_s_wait_asynccnt)
#define WAIT_ASYNC(n) __builtin_amdgcn_s_wait_asynccnt(n)
#endif
#endif
#ifndef WAIT_ASYNC
#define WAIT_ASYNC(n) asm volatile("s_wait_asynccnt %0" :: "n"(n) : "memory")
#endif

// ===========================================================================
// f32 -> bf16 convert (grid-stride)
// ===========================================================================
__global__ __launch_bounds__(256)
void k_cvt_bf16(const float* __restrict__ x, bf16* __restrict__ y, size_t n)
{
    size_t i  = (size_t)blockIdx.x * blockDim.x + threadIdx.x;
    size_t st = (size_t)gridDim.x * blockDim.x;
    for (; i < n; i += st) y[i] = (bf16)x[i];
}

// ===========================================================================
// Tiled WMMA GEMM:  Y[R,Dout] = X[R,K](bf16) * W[Dout,K](bf16)^T + bias
// 128x128 block, 4 waves, each wave 64x64 via 16 accumulators.
// Tiles staged by async global->LDS DMA, double-buffered.
// ===========================================================================
#define LDSROW 40   // halfs per LDS row (80B = 20-bank step, b128-conflict-free)

__global__ __launch_bounds__(128, 1)
void k_gemm_bf16(const bf16* __restrict__ X, const bf16* __restrict__ W,
                 const float* __restrict__ bias, float* __restrict__ Y,
                 int R, int K, int Dout)
{
    __shared__ bf16 As[2][128 * LDSROW];
    __shared__ bf16 Bs[2][128 * LDSROW];

    const int tid  = threadIdx.x;
    const int lane = tid & 31;
    const int wave = tid >> 5;
    const int lh   = lane >> 4;     // half-wave (selects K/M sub-chunks)
    const int lm   = lane & 15;
    const int rblk = (int)blockIdx.y * 128;
    const int cblk = (int)blockIdx.x * 128;
    const int nk   = K >> 5;

    v8f acc[4][4] = {};

#if USE_ASYNC_LDS
    // Per k-step each wave issues 8 async b128 copies (4 for A rows, 4 for B
    // rows). Lane -> (row-in-8-group, 16B chunk of the 64B row).
    const int lrow8 = lane >> 2;    // 0..7
    const int lchk  = lane & 3;     // 0..3
    const unsigned long long xbase = (unsigned long long)(uintptr_t)X;
    const unsigned long long wbase = (unsigned long long)(uintptr_t)W;

    auto issue_tiles = [&](int buf, int k0) {
        const unsigned ldsA = lds_off(&As[buf][0]);
        const unsigned ldsB = lds_off(&Bs[buf][0]);
        #pragma unroll
        for (int q = 0; q < 4; ++q) {
            const int row  = wave * 32 + q * 8 + lrow8;          // tile row 0..127
            const unsigned la = ldsA + (unsigned)row * (LDSROW * 2) + lchk * 16;
            const unsigned lb = ldsB + (unsigned)row * (LDSROW * 2) + lchk * 16;
            const int grow = rblk + row;                          // X row
            const int gcol = cblk + row;                          // W row (= out col)
            if (grow < R) {
                const unsigned goff = (unsigned)(((size_t)grow * K + k0) * 2) + lchk * 16;
                asm volatile("global_load_async_to_lds_b128 %0, %1, %2"
                             :: "v"(la), "v"(goff), "s"(xbase) : "memory");
            }
            if (gcol < Dout) {
                const unsigned goff = (unsigned)(((size_t)gcol * K + k0) * 2) + lchk * 16;
                asm volatile("global_load_async_to_lds_b128 %0, %1, %2"
                             :: "v"(lb), "v"(goff), "s"(wbase) : "memory");
            }
        }
    };

    issue_tiles(0, 0);
#else
    const int arow = rblk + tid;
    const int brow = cblk + tid;
    U128 ar[4], br[4];
    auto ldA = [&](int k0) {
        if (arow < R) {
            const U128* p = (const U128*)(X + (size_t)arow * K + k0);
            ar[0] = p[0]; ar[1] = p[1]; ar[2] = p[2]; ar[3] = p[3];
        } else {
            U128 z = {{0u,0u,0u,0u}};
            ar[0] = z; ar[1] = z; ar[2] = z; ar[3] = z;
        }
    };
    auto ldB = [&](int k0) {
        if (brow < Dout) {
            const U128* p = (const U128*)(W + (size_t)brow * K + k0);
            br[0] = p[0]; br[1] = p[1]; br[2] = p[2]; br[3] = p[3];
        } else {
            U128 z = {{0u,0u,0u,0u}};
            br[0] = z; br[1] = z; br[2] = z; br[3] = z;
        }
    };
    ldA(0); ldB(0);
#endif

    for (int it = 0; it < nk; ++it) {
#if USE_ASYNC_LDS
        if (it + 1 < nk) {
            issue_tiles((it + 1) & 1, (it + 1) * 32);
            WAIT_ASYNC(8);   // in-order => current buffer's 8 copies have landed
        } else {
            WAIT_ASYNC(0);
        }
        __syncthreads();
        const bf16* Acur = &As[it & 1][0];
        const bf16* Bcur = &Bs[it & 1][0];
#else
        __syncthreads();
        *(U128*)(&As[0][0] + tid * LDSROW +  0) = ar[0];
        *(U128*)(&As[0][0] + tid * LDSROW +  8) = ar[1];
        *(U128*)(&As[0][0] + tid * LDSROW + 16) = ar[2];
        *(U128*)(&As[0][0] + tid * LDSROW + 24) = ar[3];
        *(U128*)(&Bs[0][0] + tid * LDSROW +  0) = br[0];
        *(U128*)(&Bs[0][0] + tid * LDSROW +  8) = br[1];
        *(U128*)(&Bs[0][0] + tid * LDSROW + 16) = br[2];
        *(U128*)(&Bs[0][0] + tid * LDSROW + 24) = br[3];
        __syncthreads();
        if (it + 1 < nk) { ldA((it + 1) * 32); ldB((it + 1) * 32); }
        const bf16* Acur = &As[0][0];
        const bf16* Bcur = &Bs[0][0];
#endif

        v16bf af[4], bfr[4];
        // A 16x32 bf16 per-lane layout (ISA 7.12.2): lanes 0-15 halfs 0..7 -> K 0..7,
        // halfs 8..15 -> K 16..23; lanes 16-31 shifted by +8 in K.
        #pragma unroll
        for (int mt = 0; mt < 4; ++mt) {
            const bf16* ap = Acur + ((wave >> 1) * 64 + mt * 16 + lm) * LDSROW;
            v8bf lo = *(const v8bf*)(ap + 8 * lh);
            v8bf hi = *(const v8bf*)(ap + 16 + 8 * lh);
            af[mt] = __builtin_shufflevector(lo, hi, 0,1,2,3,4,5,6,7,8,9,10,11,12,13,14,15);
        }
        // B 32x16: lane n = lm, same K layout as A: halfs 0..7 -> K 8*lh.., halfs 8..15 -> K 16+8*lh...
        #pragma unroll
        for (int nt = 0; nt < 4; ++nt) {
            const bf16* bp = Bcur + ((wave & 1) * 64 + nt * 16 + lm) * LDSROW;
            v8bf lo = *(const v8bf*)(bp + 8 * lh);
            v8bf hi = *(const v8bf*)(bp + 16 + 8 * lh);
            bfr[nt] = __builtin_shufflevector(lo, hi, 0,1,2,3,4,5,6,7,8,9,10,11,12,13,14,15);
        }
        #pragma unroll
        for (int mt = 0; mt < 4; ++mt)
            #pragma unroll
            for (int nt = 0; nt < 4; ++nt)
                acc[mt][nt] = __builtin_amdgcn_wmma_f32_16x16x32_bf16(
                    false, af[mt], false, bfr[nt], (short)0, acc[mt][nt], false, false);

        __syncthreads();
    }

    // Epilogue: C/D 16x16 layout -> lane n = lm, VGPR i -> m = i + 8*lh.
    #pragma unroll
    for (int nt = 0; nt < 4; ++nt) {
        const int col = cblk + (wave & 1) * 64 + nt * 16 + lm;
        if (col >= Dout) continue;
        const float bv = bias ? bias[col] : 0.f;
        #pragma unroll
        for (int mt = 0; mt < 4; ++mt) {
            #pragma unroll
            for (int i = 0; i < 8; ++i) {
                const int row = rblk + (wave >> 1) * 64 + mt * 16 + i + 8 * lh;
                if (row < R) Y[(size_t)row * Dout + col] = acc[mt][nt][i] + bv;
            }
        }
    }
}

// ===========================================================================
// Row LayerNorm (+ optional exact-erf GELU), one block per row.
// ===========================================================================
template <bool GELU, typename OutT>
__global__ __launch_bounds__(256)
void k_ln_act(const float* __restrict__ X, const float* __restrict__ g,
              const float* __restrict__ be, OutT* __restrict__ Y, int C)
{
    const int row = blockIdx.x, tid = threadIdx.x;
    const float* x = X + (size_t)row * C;
    OutT* y = Y + (size_t)row * C;
    __shared__ float rs[256], rq[256];
    float s = 0.f, q = 0.f;
    for (int c = tid; c < C; c += 256) { float v = x[c]; s += v; q += v * v; }
    rs[tid] = s; rq[tid] = q; __syncthreads();
    for (int off = 128; off > 0; off >>= 1) {
        if (tid < off) { rs[tid] += rs[tid + off]; rq[tid] += rq[tid + off]; }
        __syncthreads();
    }
    const float mu   = rs[0] / C;
    const float var  = rq[0] / C - mu * mu;
    const float rstd = rsqrtf(var + 1e-5f);
    for (int c = tid; c < C; c += 256) {
        float v = (x[c] - mu) * rstd * g[c] + be[c];
        if (GELU) v = 0.5f * v * (1.f + erff(v * 0.70710678118654752f));
        y[c] = (OutT)v;
    }
}

// ===========================================================================
// Naive small linear: Y[r,d] = X[r,:] . W[d,:] + b[d]  (R tiny: 1 or 8)
// ===========================================================================
__global__ __launch_bounds__(256)
void k_linear_rows(const float* __restrict__ X, const float* __restrict__ W,
                   const float* __restrict__ bias, float* __restrict__ Y,
                   int K, int Dout)
{
    const int d = blockIdx.x * 256 + threadIdx.x;
    const int r = blockIdx.y;
    if (d >= Dout) return;
    const float* x = X + (size_t)r * K;
    const float* w = W + (size_t)d * K;
    float s = bias ? bias[d] : 0.f;
    for (int k = 0; k < K; ++k) s += x[k] * w[k];
    Y[(size_t)r * Dout + d] = s;
}

// ===========================================================================
// Single-query masked MHA: Q[b? or shared][D], K/V f32 [B,L,D].
// Key j masked iff j>=maskShift && mask[b][(j-maskShift)].
// ===========================================================================
__global__ __launch_bounds__(256)
void k_attn1q(const float* __restrict__ Q, int qstride,
              const float* __restrict__ Km, const float* __restrict__ Vm,
              const int* __restrict__ mask, int maskShift,
              float scale, int L, int Dfull, int hd,
              float* __restrict__ O)
{
    const int h = blockIdx.x, b = blockIdx.y, tid = threadIdx.x;
    __shared__ float qs[256];
    __shared__ float red[256];
    __shared__ F4    r4[256];
    for (int c = tid; c < hd; c += 256) qs[c] = Q[(size_t)b * qstride + (size_t)h * hd + c];
    __syncthreads();

    float sv[17];   // ceil(4097/256) = 17 keys/thread max
    int   cnt = 0;
    float lmax = -3.0e38f;
    for (int j = tid; j < L; j += 256) {
        float s = -1.0e30f;
        const bool msk = (mask != nullptr) && (j >= maskShift) &&
                         (mask[(size_t)b * (L - maskShift) + (j - maskShift)] != 0);
        if (!msk) {
            const float* kr = Km + ((size_t)b * L + j) * Dfull + (size_t)h * hd;
            float a = 0.f;
            for (int c = 0; c < hd; ++c) a += qs[c] * kr[c];
            s = a * scale;
        }
        sv[cnt++] = s;
        lmax = fmaxf(lmax, s);
    }
    red[tid] = lmax; __syncthreads();
    for (int off = 128; off > 0; off >>= 1) {
        if (tid < off) red[tid] = fmaxf(red[tid], red[tid + off]);
        __syncthreads();
    }
    const float gmax = red[0]; __syncthreads();
    float lsum = 0.f;
    for (int i = 0; i < cnt; ++i) { float e = __expf(sv[i] - gmax); sv[i] = e; lsum += e; }
    red[tid] = lsum; __syncthreads();
    for (int off = 128; off > 0; off >>= 1) {
        if (tid < off) red[tid] += red[tid + off];
        __syncthreads();
    }
    const float inv = 1.f / red[0]; __syncthreads();

    for (int dc = 0; dc < hd; dc += 4) {
        F4 a = {0.f, 0.f, 0.f, 0.f};
        int idx = 0;
        for (int j = tid; j < L; j += 256) {
            const float w = sv[idx++] * inv;
            const float* vr = Vm + ((size_t)b * L + j) * Dfull + (size_t)h * hd + dc;
            a.x += w * vr[0]; a.y += w * vr[1]; a.z += w * vr[2]; a.w += w * vr[3];
        }
        r4[tid] = a; __syncthreads();
        for (int off = 128; off > 0; off >>= 1) {
            if (tid < off) {
                F4 o = r4[tid + off];
                r4[tid].x += o.x; r4[tid].y += o.y; r4[tid].z += o.z; r4[tid].w += o.w;
            }
            __syncthreads();
        }
        if (tid == 0) {
            float* op = O + (size_t)b * Dfull + (size_t)h * hd + dc;
            op[0] = r4[0].x; op[1] = r4[0].y; op[2] = r4[0].z; op[3] = r4[0].w;
        }
        __syncthreads();
    }
}

// ===========================================================================
// Cosine similarity + score prep: sim[b,n] = clip((cos+1)/2, 0, 1)
// ===========================================================================
__global__ __launch_bounds__(256)
void k_sim(const float* __restrict__ fc, const float* __restrict__ cc,
           const int* __restrict__ fmask, float* __restrict__ sim_out,
           float* __restrict__ sscore, float* __restrict__ uscore, int n)
{
    const int b = blockIdx.y;
    const int i = blockIdx.x * 256 + threadIdx.x;
    __shared__ float ccs[64];
    __shared__ float ccn;
    if (threadIdx.x < 64) ccs[threadIdx.x] = cc[b * 64 + threadIdx.x];
    __syncthreads();
    if (threadIdx.x == 0) {
        float s = 0.f;
        for (int c = 0; c < 64; ++c) s += ccs[c] * ccs[c];
        ccn = sqrtf(s);
    }
    __syncthreads();
    if (i < n) {
        const float* f = fc + ((size_t)b * n + i) * 64;
        float nf = 0.f, dt = 0.f;
        for (int c = 0; c < 64; ++c) { nf += f[c] * f[c]; dt += f[c] * ccs[c]; }
        const float cosv = dt / (sqrtf(nf) * ccn);
        const float sm = fminf(fmaxf((cosv + 1.f) * 0.5f, 0.f), 1.f);
        sim_out[(size_t)b * n + i] = sm;
        const bool m = fmask[(size_t)b * n + i] != 0;
        sscore[(size_t)b * n + i] = m ? -1.f : sm;
        uscore[(size_t)b * n + i] = m ?  2.f : sm;
    }
}

// ===========================================================================
// Top-k (k=128) by iterative argmax; jax tie-break = lowest index first.
// sign=+1 -> largest, sign=-1 -> smallest. One block per batch, n = 4096.
// ===========================================================================
__global__ __launch_bounds__(256)
void k_topk(const float* __restrict__ score, int n, int k, float sign,
            int* __restrict__ idx)
{
    __shared__ float s[4096];
    __shared__ float rv[256];
    __shared__ int   ri[256];
    const int b = blockIdx.x, tid = threadIdx.x;
    for (int i = tid; i < n; i += 256) s[i] = sign * score[(size_t)b * n + i];
    __syncthreads();
    for (int t = 0; t < k; ++t) {
        float bv = -3.0e38f; int bi = n;
        for (int i = tid; i < n; i += 256) {
            const float v = s[i];
            if (v > bv) { bv = v; bi = i; }
        }
        rv[tid] = bv; ri[tid] = bi; __syncthreads();
        for (int off = 128; off > 0; off >>= 1) {
            if (tid < off) {
                const float ov = rv[tid + off]; const int oi = ri[tid + off];
                if (ov > rv[tid] || (ov == rv[tid] && oi < ri[tid])) { rv[tid] = ov; ri[tid] = oi; }
            }
            __syncthreads();
        }
        if (tid == 0) { idx[(size_t)b * k + t] = ri[0]; s[ri[0]] = -3.4e38f; }
        __syncthreads();
    }
}

// ===========================================================================
// Gather rows to bf16 patches: out[b,t,:] = bf16(src[b, idx[b,t], :])
// ===========================================================================
__global__ __launch_bounds__(256)
void k_gather_bf16(const float* __restrict__ src, const int* __restrict__ idx,
                   bf16* __restrict__ out, int n, int d, int tok)
{
    const int t = blockIdx.x, b = blockIdx.y;
    const int row = idx[(size_t)b * tok + t];
    const float* s = src + ((size_t)b * n + row) * d;
    bf16* o = out + ((size_t)b * tok + t) * d;
    for (int c = threadIdx.x; c < d; c += 256) o[c] = (bf16)s[c];
}

// ===========================================================================
// Build concat([global, feats]) rows: out[b*(n+1)+j] = j==0 ? glob[b] : feats[b,j-1]
// ===========================================================================
__global__ __launch_bounds__(256)
void k_concat_kvf(const float* __restrict__ glob, const float* __restrict__ feats,
                  float* __restrict__ out, int n, int d)
{
    const int j = blockIdx.x, b = blockIdx.y;
    const float* s = (j == 0) ? glob + (size_t)b * d
                              : feats + ((size_t)b * n + (j - 1)) * d;
    float* o = out + ((size_t)b * (n + 1) + j) * d;
    for (int c = threadIdx.x; c < d; c += 256) o[c] = s[c];
}

__global__ __launch_bounds__(256)
void k_add(float* __restrict__ y, const float* __restrict__ x, int n)
{
    const int i = blockIdx.x * 256 + threadIdx.x;
    if (i < n) y[i] += x[i];
}

__global__ __launch_bounds__(256)
void k_copy(const float* __restrict__ x, float* __restrict__ y, int n)
{
    const int i = blockIdx.x * 256 + threadIdx.x;
    if (i < n) y[i] = x[i];
}

// ===========================================================================
// Host orchestration
// ===========================================================================
struct MlpP { const float *w1,*b1,*g1,*be1,*w2,*b2,*g2,*be2,*w3,*b3; };
struct MhaP { const float *wq,*bq,*wk,*bk,*wv,*bv,*wo,*bo; };

// jax pytree leaves = sorted dict keys (recursive):
// mlp leaf order: b1 b2 b3 be1 be2 g1 g2 w1 w2 w3
static MlpP mlp_at(void* const* d_in, int base) {
    MlpP p;
    p.b1 = (const float*)d_in[base+0]; p.b2 = (const float*)d_in[base+1];
    p.b3 = (const float*)d_in[base+2]; p.be1= (const float*)d_in[base+3];
    p.be2= (const float*)d_in[base+4]; p.g1 = (const float*)d_in[base+5];
    p.g2 = (const float*)d_in[base+6]; p.w1 = (const float*)d_in[base+7];
    p.w2 = (const float*)d_in[base+8]; p.w3 = (const float*)d_in[base+9];
    return p;
}
// mha leaf order: bk bo bq bv wk wo wq wv
static MhaP mha_at(void* const* d_in, int base) {
    MhaP p;
    p.bk = (const float*)d_in[base+0]; p.bo = (const float*)d_in[base+1];
    p.bq = (const float*)d_in[base+2]; p.bv = (const float*)d_in[base+3];
    p.wk = (const float*)d_in[base+4]; p.wo = (const float*)d_in[base+5];
    p.wq = (const float*)d_in[base+6]; p.wv = (const float*)d_in[base+7];
    return p;
}

extern "C" void kernel_launch(void* const* d_in, const int* in_sizes, int n_in,
                              void* d_out, int out_size, void* d_ws, size_t ws_size,
                              hipStream_t stream)
{
    (void)in_sizes; (void)n_in; (void)out_size; (void)ws_size;

    // ---- inputs (setup_inputs dict order; params pytree-flattened) ----
    const float* feats = (const float*)d_in[0];   // [B,N,D]
    const float* glob  = (const float*)d_in[1];   // [B,1,D]
    const float* cross = (const float*)d_in[2];   // [B,M,D]
    const int*   fmask = (const int*)d_in[3];     // [B,N]
    const int*   cmask = (const int*)d_in[4];     // [B,M]
    const float* cls_cross = (const float*)d_in[5];
    const float* cls_feat  = (const float*)d_in[6];
    MlpP cs = mlp_at(d_in, 7);    // consist_feat
    MlpP ce = mlp_at(d_in, 17);   // cross_enc
    MlpP fe = mlp_at(d_in, 27);   // feat_enc
    const float* lnc_b = (const float*)d_in[37];
    const float* lnc_g = (const float*)d_in[38];
    const float* lnf_b = (const float*)d_in[39];
    const float* lnf_g = (const float*)d_in[40];
    MhaP m1 = mha_at(d_in, 41);
    MhaP m2 = mha_at(d_in, 49);
    MhaP mc = mha_at(d_in, 57);   // mha_cross
    MhaP mf = mha_at(d_in, 65);   // mha_feat
    MlpP l1 = mlp_at(d_in, 73);   // mlp1
    MlpP l2 = mlp_at(d_in, 83);   // mlp2

    float* outp = (float*)d_out;
    float* out_fa   = outp;                 // [B,1,D]   8192
    float* out_sim  = outp + 8192;          // [B,N]     32768
    float* out_fcns = outp + 8192 + 32768;  // [B,N,64]  2097152

    // ---- workspace bump allocator ----
    size_t off = 0;
    auto A = [&](size_t bytes) -> void* {
        void* p = (char*)d_ws + off;
        off = (off + bytes + 255) & ~(size_t)255;
        return p;
    };
    const size_t RMAX = (size_t)RMAXm;

    bf16* fe_w1b = (bf16*)A((size_t)Hm*Dm*2);
    bf16* fe_w2b = (bf16*)A((size_t)Hm*Hm*2);
    bf16* fe_w3b = (bf16*)A((size_t)Dm*Hm*2);
    bf16* ce_w1b = (bf16*)A((size_t)Hm*Dm*2);
    bf16* ce_w2b = (bf16*)A((size_t)Hm*Hm*2);
    bf16* ce_w3b = (bf16*)A((size_t)Dm*Hm*2);
    bf16* cs_w1b = (bf16*)A((size_t)256*Dm*2);
    bf16* cs_w2b = (bf16*)A((size_t)128*256*2);
    bf16* cs_w3b = (bf16*)A((size_t)64*128*2);
    bf16* wkc = (bf16*)A((size_t)Dm*Dm*2);
    bf16* wvc = (bf16*)A((size_t)Dm*Dm*2);
    bf16* wkf = (bf16*)A((size_t)Dm*Dm*2);
    bf16* wvf = (bf16*)A((size_t)Dm*Dm*2);
    bf16* wk1 = (bf16*)A((size_t)Dm*Dm*2);
    bf16* wv1 = (bf16*)A((size_t)Dm*Dm*2);
    bf16* wk2 = (bf16*)A((size_t)Dm*Dm*2);
    bf16* wv2 = (bf16*)A((size_t)Dm*Dm*2);

    bf16*  actA = (bf16*)A(RMAX*Hm*2);
    bf16*  actB = (bf16*)A(RMAX*Hm*2);
    float* gout = (float*)A(RMAX*Hm*4);
    float* kbuf = gout;                 // alias: K/V reuse GEMM scratch
    float* vbuf = gout + RMAX*Dm;       //  after the big MLP phases are done
    float* fenc   = (float*)A((size_t)Bm*Nm*Dm*4);
    float* cenc   = (float*)A((size_t)Bm*Mm*Dm*4);
    float* kvrows = (float*)A(RMAX*Dm*4);
    float* sscore = (float*)A((size_t)Bm*Nm*4);
    float* uscore = (float*)A((size_t)Bm*Nm*4);
    int*   idxS   = (int*)A((size_t)Bm*TOKm*4);
    int*   idxU   = (int*)A((size_t)Bm*TOKm*4);
    bf16*  simp   = (bf16*)A((size_t)Bm*TOKm*Dm*2);
    bf16*  unsimp = (bf16*)A((size_t)Bm*TOKm*Dm*2);
    float* cc     = (float*)A((size_t)Bm*64*4);
    float* qln    = (float*)A((size_t)Dm*4);
    float* qvec   = (float*)A((size_t)Dm*4);
    float* qrows  = (float*)A((size_t)Bm*Dm*4);
    float* ohead  = (float*)A((size_t)Bm*Dm*4);
    float* facr   = (float*)A((size_t)Bm*Dm*4);
    float* fa     = (float*)A((size_t)Bm*Dm*4);
    float* mo     = (float*)A((size_t)Bm*Dm*4);
    float* mres   = (float*)A((size_t)Bm*Dm*4);
    float* tsa    = (float*)A((size_t)Bm*Hm*4);
    float* tsb    = (float*)A((size_t)Bm*Hm*4);

    // ---- launch helpers ----
    auto cvt = [&](const float* x, bf16* y, size_t n) {
        size_t blocks = (n + 255) / 256; if (blocks > 4096) blocks = 4096;
        k_cvt_bf16<<<(unsigned)blocks, 256, 0, stream>>>(x, y, n);
    };
    auto gemm = [&](const bf16* X, const bf16* W, const float* bias, float* Y,
                    int R, int K, int Dout) {
        dim3 g((Dout + 127) / 128, (R + 127) / 128);
        k_gemm_bf16<<<g, 128, 0, stream>>>(X, W, bias, Y, R, K, Dout);
    };
    auto ln_bf16 = [&](const float* X, const float* g, const float* b, bf16* Y, int R, int C) {
        k_ln_act<false, bf16><<<R, 256, 0, stream>>>(X, g, b, Y, C);
    };
    auto ln_gelu_bf16 = [&](const float* X, const float* g, const float* b, bf16* Y, int R, int C) {
        k_ln_act<true, bf16><<<R, 256, 0, stream>>>(X, g, b, Y, C);
    };
    auto ln_f32 = [&](const float* X, const float* g, const float* b, float* Y, int R, int C) {
        k_ln_act<false, float><<<R, 256, 0, stream>>>(X, g, b, Y, C);
    };
    auto ln_gelu_f32 = [&](const float* X, const float* g, const float* b, float* Y, int R, int C) {
        k_ln_act<true, float><<<R, 256, 0, stream>>>(X, g, b, Y, C);
    };
    auto lin = [&](const float* X, const float* W, const float* bias, float* Y,
                   int rows, int K, int Dout) {
        dim3 g((Dout + 255) / 256, rows);
        k_linear_rows<<<g, 256, 0, stream>>>(X, W, bias, Y, K, Dout);
    };
    auto big_mlp = [&](const float* Xin, int rows, const MlpP& p,
                       const bf16* w1b, const bf16* w2b, const bf16* w3b,
                       int d0, int d1, int d2, int d3, float* Yout) {
        cvt(Xin, actA, (size_t)rows * d0);
        gemm(actA, w1b, p.b1, gout, rows, d0, d1);
        ln_gelu_bf16(gout, p.g1, p.be1, actB, rows, d1);
        gemm(actB, w2b, p.b2, gout, rows, d1, d2);
        ln_gelu_bf16(gout, p.g2, p.be2, actA, rows, d2);
        gemm(actA, w3b, p.b3, Yout, rows, d2, d3);
    };
    auto small_mlp = [&](const float* Xin, const MlpP& p,
                         int d0, int d1, int d2, int d3, float* Yout) {
        lin(Xin, p.w1, p.b1, tsa, Bm, d0, d1);
        ln_gelu_f32(tsa, p.g1, p.be1, tsb, Bm, d1);
        lin(tsb, p.w2, p.b2, tsa, Bm, d1, d2);
        ln_gelu_f32(tsa, p.g2, p.be2, tsb, Bm, d2);
        lin(tsb, p.w3, p.b3, Yout, Bm, d2, d3);
    };

    const float scale8 = 0.08838834765f;   // 1/sqrt(128)
    const float scale4 = 0.0625f;          // 1/sqrt(256)

    // ---- phase 0: weight conversions to bf16 ----
    cvt(fe.w1, fe_w1b, (size_t)Hm*Dm);  cvt(fe.w2, fe_w2b, (size_t)Hm*Hm);  cvt(fe.w3, fe_w3b, (size_t)Dm*Hm);
    cvt(ce.w1, ce_w1b, (size_t)Hm*Dm);  cvt(ce.w2, ce_w2b, (size_t)Hm*Hm);  cvt(ce.w3, ce_w3b, (size_t)Dm*Hm);
    cvt(cs.w1, cs_w1b, (size_t)256*Dm); cvt(cs.w2, cs_w2b, (size_t)128*256); cvt(cs.w3, cs_w3b, (size_t)64*128);
    cvt(mc.wk, wkc, (size_t)Dm*Dm); cvt(mc.wv, wvc, (size_t)Dm*Dm);
    cvt(mf.wk, wkf, (size_t)Dm*Dm); cvt(mf.wv, wvf, (size_t)Dm*Dm);
    cvt(m1.wk, wk1, (size_t)Dm*Dm); cvt(m1.wv, wv1, (size_t)Dm*Dm);
    cvt(m2.wk, wk2, (size_t)Dm*Dm); cvt(m2.wv, wv2, (size_t)Dm*Dm);

    // ---- phase 1: big encoder MLPs (gout scratch) ----
    big_mlp(feats, Bm*Nm, fe, fe_w1b, fe_w2b, fe_w3b, Dm, Hm, Hm, Dm, fenc);
    big_mlp(cross, Bm*Mm, ce, ce_w1b, ce_w2b, ce_w3b, Dm, Hm, Hm, Dm, cenc);
    // consist MLP on feats -> directly into d_out region 3
    big_mlp(fenc, Bm*Nm, cs, cs_w1b, cs_w2b, cs_w3b, Dm, 256, 128, 64, out_fcns);

    // ---- phase 2: mha_cross (cls query over LN(cross_enc)) ----
    ln_bf16(cenc, lnc_g, lnc_b, actA, Bm*Mm, Dm);
    gemm(actA, wkc, mc.bk, kbuf, Bm*Mm, Dm, Dm);
    gemm(actA, wvc, mc.bv, vbuf, Bm*Mm, Dm, Dm);
    ln_f32(cls_cross, lnc_g, lnc_b, qln, 1, Dm);
    lin(qln, mc.wq, mc.bq, qvec, 1, Dm, Dm);
    k_attn1q<<<dim3(NHm, Bm), 256, 0, stream>>>(qvec, 0, kbuf, vbuf, cmask, 0,
                                                scale8, Mm, Dm, Dm/NHm, ohead);
    lin(ohead, mc.wo, mc.bo, facr, Bm, Dm, Dm);

    // ---- phase 3: consist(cross) + similarity + routing scores ----
    small_mlp(facr, cs, Dm, 256, 128, 64, cc);
    {
        dim3 g((Nm + 255) / 256, Bm);
        k_sim<<<g, 256, 0, stream>>>(out_fcns, cc, fmask, out_sim, sscore, uscore, Nm);
    }
    k_topk<<<Bm, 256, 0, stream>>>(sscore, Nm, TOKm,  1.f, idxS);
    k_topk<<<Bm, 256, 0, stream>>>(uscore, Nm, TOKm, -1.f, idxU);
    k_gather_bf16<<<dim3(TOKm, Bm), 256, 0, stream>>>(fenc, idxS, simp,   Nm, Dm, TOKm);
    k_gather_bf16<<<dim3(TOKm, Bm), 256, 0, stream>>>(fenc, idxU, unsimp, Nm, Dm, TOKm);

    // ---- phase 4: mha_feat (cls query over LN(concat[global, feats])) ----
    k_concat_kvf<<<dim3(Nm + 1, Bm), 256, 0, stream>>>(glob, fenc, kvrows, Nm, Dm);
    ln_bf16(kvrows, lnf_g, lnf_b, actA, Bm*(Nm+1), Dm);
    gemm(actA, wkf, mf.bk, kbuf, Bm*(Nm+1), Dm, Dm);
    gemm(actA, wvf, mf.bv, vbuf, Bm*(Nm+1), Dm, Dm);
    ln_f32(cls_feat, lnf_g, lnf_b, qln, 1, Dm);
    lin(qln, mf.wq, mf.bq, qvec, 1, Dm, Dm);
    k_attn1q<<<dim3(NHm, Bm), 256, 0, stream>>>(qvec, 0, kbuf, vbuf, fmask, 1,
                                                scale8, Nm + 1, Dm, Dm/NHm, ohead);
    lin(ohead, mf.wo, mf.bo, fa, Bm, Dm, Dm);

    // ---- phase 5: mha1/mlp1 over sim patches (nh=4), residual add ----
    gemm(simp, wk1, m1.bk, kbuf, Bm*TOKm, Dm, Dm);
    gemm(simp, wv1, m1.bv, vbuf, Bm*TOKm, Dm, Dm);
    lin(fa, m1.wq, m1.bq, qrows, Bm, Dm, Dm);
    k_attn1q<<<dim3(4, Bm), 256, 0, stream>>>(qrows, Dm, kbuf, vbuf, nullptr, 0,
                                              scale4, TOKm, Dm, 256, ohead);
    lin(ohead, m1.wo, m1.bo, mo, Bm, Dm, Dm);
    small_mlp(mo, l1, Dm, Hm, Hm, Dm, mres);
    k_add<<<(Bm*Dm + 255) / 256, 256, 0, stream>>>(fa, mres, Bm*Dm);

    // ---- phase 6: mha2/mlp2 over unsim patches (nh=4), residual add ----
    gemm(unsimp, wk2, m2.bk, kbuf, Bm*TOKm, Dm, Dm);
    gemm(unsimp, wv2, m2.bv, vbuf, Bm*TOKm, Dm, Dm);
    lin(fa, m2.wq, m2.bq, qrows, Bm, Dm, Dm);
    k_attn1q<<<dim3(4, Bm), 256, 0, stream>>>(qrows, Dm, kbuf, vbuf, nullptr, 0,
                                              scale4, TOKm, Dm, 256, ohead);
    lin(ohead, m2.wo, m2.bo, mo, Bm, Dm, Dm);
    small_mlp(mo, l2, Dm, Hm, Hm, Dm, mres);
    k_add<<<(Bm*Dm + 255) / 256, 256, 0, stream>>>(fa, mres, Bm*Dm);

    // ---- final: feat_aggr -> d_out region 1 ----
    k_copy<<<(Bm*Dm + 255) / 256, 256, 0, stream>>>(fa, out_fa, Bm*Dm);
}
